// LSTM_CRF_75136157876550
// MI455X (gfx1250) — compile-verified, matrix operand order checked by reading, not run on hardware
//
#include <hip/hip_runtime.h>
#include <stdint.h>

// ---- sizes ----
#define BB 32
#define SS 512
#define DD 768
#define HH 256
#define TT 64
#define NG 1024  // 4*H

typedef __attribute__((ext_vector_type(16))) __bf16 v16bf;
typedef __attribute__((ext_vector_type(8)))  __bf16 v8bf;
typedef __attribute__((ext_vector_type(8)))  float  v8f;
typedef __bf16 bf16;

union Frag { v16bf v; v8bf h[2]; };

__device__ __forceinline__ bf16 f2bf(float f) {
  union { float f; unsigned u; } x; x.f = f;
  unsigned r = x.u + 0x7FFFu + ((x.u >> 16) & 1u);
  unsigned short hs = (unsigned short)(r >> 16);
  return __builtin_bit_cast(bf16, hs);
}
__device__ __forceinline__ float bf2f(bf16 b) {
  unsigned short hs = __builtin_bit_cast(unsigned short, b);
  union { unsigned u; float f; } x; x.u = ((unsigned)hs) << 16;
  return x.f;
}
__device__ __forceinline__ float sigf(float x) { return 1.f / (1.f + __expf(-x)); }

// ---------- f32 -> bf16 convert ----------
__global__ void f32_to_bf16_kernel(const float* __restrict__ in, bf16* __restrict__ out, int n) {
  int i = blockIdx.x * blockDim.x + threadIdx.x;
  if (i < n) out[i] = f2bf(in[i]);
}

// ---------- generic WMMA GEMM: out[M,N] = A[M,K] @ B[N,K]^T + bias (+bias2) ----------
// 128 threads = 4 waves; each wave owns one 16x16 tile.
template<int OUT_BF16, int GELU>
__global__ __launch_bounds__(128)
void gemm_bf16_wmma(const bf16* __restrict__ A, const bf16* __restrict__ Bm,
                    const float* __restrict__ bias, const float* __restrict__ bias2,
                    void* __restrict__ outv, int M, int N, int K)
{
  const int lane = threadIdx.x & 31;
  const int wave = threadIdx.x >> 5;
  const int mBase = (blockIdx.y * 4 + wave) * 16;
  const int nBase = blockIdx.x * 16;
  if (mBase >= M) return;
  const int row  = lane & 15;
  const int half = lane >> 4;
  const bf16* ap = A  + (size_t)(mBase + row) * K + half * 8;
  const bf16* bp = Bm + (size_t)(nBase + row) * K + half * 8;
  v8f acc = {};
  for (int k = 0; k < K; k += 32) {
    Frag a, b;
    a.h[0] = *(const v8bf*)(ap + k);
    a.h[1] = *(const v8bf*)(ap + k + 16);
    b.h[0] = *(const v8bf*)(bp + k);
    b.h[1] = *(const v8bf*)(bp + k + 16);
    __builtin_prefetch(bp + k + 64, 0, 1);
    acc = __builtin_amdgcn_wmma_f32_16x16x32_bf16(false, a.v, false, b.v,
                                                  (short)0, acc, false, false);
  }
  const int n = nBase + row;
  float bsum = bias ? bias[n] : 0.f;
  if (bias2) bsum += bias2[n];
  const int m0 = mBase + half * 8;
  for (int r = 0; r < 8; ++r) {
    float v = acc[r] + bsum;
    if (GELU) v = 0.5f * v * (1.f + erff(v * 0.70710678118654752f));
    size_t o = (size_t)(m0 + r) * N + n;
    if (OUT_BF16) ((bf16*)outv)[o] = f2bf(v);
    else          ((float*)outv)[o] = v;
  }
}

// ---------- BiLSTM recurrence ----------
// 4 blocks: blockIdx.x = dir*2 + mtile. Each block: 16 batches, full N=1024,
// 1024 threads = 32 waves, each wave does 2 N-tiles. h kept in LDS (bf16),
// cell state in registers (4 cells/thread) for all 512 steps.
// pre[t] slabs (32 KB) are staged into LDS with async global->LDS DMA
// (GLOBAL_LOAD_ASYNC_TO_LDS_B128, ASYNCcnt) issued one timestep ahead so the
// copy overlaps the WMMA phase.
__global__ __launch_bounds__(1024)
void lstm_recurrence(const bf16* __restrict__ pre_f, const bf16* __restrict__ pre_b,
                     const bf16* __restrict__ whh_f, const bf16* __restrict__ whh_b,
                     bf16* __restrict__ hcat)
{
  const int dir = blockIdx.x >> 1;
  const int b0  = (blockIdx.x & 1) * 16;
  const bf16* pre = dir ? pre_b : pre_f;
  const bf16* whh = dir ? whh_b : whh_f;

  __shared__ __align__(16) bf16 h_lds[16 * HH];     //  8 KB
  __shared__ __align__(16) bf16 g_lds[16 * NG];     // 32 KB
  __shared__ __align__(16) bf16 pre_lds[16 * NG];   // 32 KB (async staged)

  const int tid  = threadIdx.x;
  const int lane = tid & 31;
  const int wave = tid >> 5;
  const int row  = lane & 15;
  const int half = lane >> 4;

  // issue async DMA of pre[:, t, :] slab (16 rows x 1024 bf16) into pre_lds.
  // 2048 x b128 chunks, 2 per thread; 128 chunks per row.
  auto issue_pre = [&](int t) {
    for (int c = tid; c < 2048; c += 1024) {
      const int m   = c >> 7;
      const int off = (c & 127) * 8;
      const bf16* g = pre + ((size_t)(b0 + m) * SS + t) * NG + off;
      const uint32_t l = (uint32_t)(uintptr_t)(pre_lds + m * NG + off);
      asm volatile("global_load_async_to_lds_b128 %0, %1, off"
                   :: "v"(l), "v"((unsigned long long)(uintptr_t)g)
                   : "memory");
    }
  };

  for (int i = tid; i < 16 * HH; i += 1024) h_lds[i] = f2bf(0.f);
  float c_state[4] = {0.f, 0.f, 0.f, 0.f};
  issue_pre(dir ? (SS - 1) : 0);
  __syncthreads();

  for (int step = 0; step < SS; ++step) {
    const int t = dir ? (SS - 1 - step) : step;

    // phase 1: g = h @ Whh^T (16 x 1024), distributed 2 N-tiles per wave.
    // Overlaps with the outstanding async DMA of pre[t].
    for (int nt = wave; nt < 64; nt += 32) {
      v8f acc = {};
      const bf16* hp = h_lds + row * HH + half * 8;
      const bf16* bp = whh + (size_t)(nt * 16 + row) * HH + half * 8;
      for (int k = 0; k < HH; k += 32) {
        Frag a, b;
        a.h[0] = *(const v8bf*)(hp + k);
        a.h[1] = *(const v8bf*)(hp + k + 16);
        b.h[0] = *(const v8bf*)(bp + k);
        b.h[1] = *(const v8bf*)(bp + k + 16);
        acc = __builtin_amdgcn_wmma_f32_16x16x32_bf16(false, a.v, false, b.v,
                                                      (short)0, acc, false, false);
      }
      const int n = nt * 16 + row;
      for (int r = 0; r < 8; ++r)
        g_lds[(half * 8 + r) * NG + n] = f2bf(acc[r]);
    }
    // each wave drains its own async loads, then barrier publishes pre_lds + g_lds
    asm volatile("s_wait_asynccnt 0x0" ::: "memory");
    __syncthreads();

    // phase 2: gates + state update (4 cells per thread), pre from LDS
    for (int q = 0; q < 4; ++q) {
      const int idx = tid + q * 1024;
      const int m = idx >> 8;
      const int n = idx & 255;
      const float gi = bf2f(g_lds[m * NG + n          ]) + bf2f(pre_lds[m * NG + n          ]);
      const float gf = bf2f(g_lds[m * NG + n + HH     ]) + bf2f(pre_lds[m * NG + n + HH     ]);
      const float gg = bf2f(g_lds[m * NG + n + 2 * HH ]) + bf2f(pre_lds[m * NG + n + 2 * HH ]);
      const float go = bf2f(g_lds[m * NG + n + 3 * HH ]) + bf2f(pre_lds[m * NG + n + 3 * HH ]);
      float c = sigf(gf) * c_state[q] + sigf(gi) * tanhf(gg);
      c_state[q] = c;
      const float h = sigf(go) * tanhf(c);
      h_lds[m * HH + n] = f2bf(h);
      hcat[((size_t)(b0 + m) * SS + t) * (2 * HH) + dir * HH + n] = f2bf(h);
    }
    __syncthreads();
    // prefetch next timestep's pre slab (overlaps next phase 1)
    if (step + 1 < SS) issue_pre(dir ? (SS - 2 - step) : (step + 1));
  }
}

// ---------- CRF log-likelihood: one block (64 threads) per batch ----------
__global__ __launch_bounds__(64)
void crf_llh_kernel(const float* __restrict__ emis, const int* __restrict__ tags,
                    const int* __restrict__ mask, const float* __restrict__ start_t,
                    const float* __restrict__ end_t, const float* __restrict__ trans,
                    float* __restrict__ llh, float* __restrict__ ntok)
{
  const int b = blockIdx.x;
  const int j = threadIdx.x;            // tag index 0..63
  __shared__ float trans_s[TT * TT];    // 16 KB
  __shared__ float alpha[TT];
  __shared__ float red[TT], red2[TT];
  __shared__ float score_sh;

  for (int i = j; i < TT * TT; i += TT) trans_s[i] = trans[i];
  alpha[j] = start_t[j] + emis[(size_t)b * SS * TT + j];

  // numerator score + token count (parallel over time, tree-reduced)
  float sc = 0.f, mk = 0.f;
  for (int t = j; t < SS; t += TT) mk += (mask[b * SS + t] != 0) ? 1.f : 0.f;
  for (int t = 1 + j; t < SS; t += TT) {
    if (mask[b * SS + t]) {
      const int tp = tags[b * SS + t - 1];
      const int tc = tags[b * SS + t];
      sc += trans[tp * TT + tc] + emis[((size_t)b * SS + t) * TT + tc];
    }
  }
  red[j] = sc; red2[j] = mk;
  __syncthreads();
  for (int o = 32; o > 0; o >>= 1) {
    if (j < o) { red[j] += red[j + o]; red2[j] += red2[j + o]; }
    __syncthreads();
  }
  if (j == 0) {
    const float total_mk = red2[0];
    const int seq_end = (int)total_mk - 1;
    const int t0 = tags[b * SS];
    const int lastt = tags[b * SS + seq_end];
    score_sh = red[0] + start_t[t0] + emis[(size_t)b * SS * TT + t0] + end_t[lastt];
    ntok[b] = total_mk;
  }
  __syncthreads();

  // forward algorithm (512-step logsumexp scan)
  for (int t = 1; t < SS; ++t) {
    const float em_j = emis[((size_t)b * SS + t) * TT + j];
    float mx = -3.4e38f;
    for (int i = 0; i < TT; ++i) mx = fmaxf(mx, alpha[i] + trans_s[i * TT + j]);
    float s = 0.f;
    for (int i = 0; i < TT; ++i) s += __expf(alpha[i] + trans_s[i * TT + j] - mx);
    const float nxt = mx + __logf(s) + em_j;
    const int m = mask[b * SS + t];
    __syncthreads();
    if (m) alpha[j] = nxt;
    __syncthreads();
  }

  // log partition
  const float val = alpha[j] + end_t[j];
  red[j] = val; __syncthreads();
  for (int o = 32; o > 0; o >>= 1) { if (j < o) red[j] = fmaxf(red[j], red[j + o]); __syncthreads(); }
  const float mx = red[0]; __syncthreads();
  red[j] = __expf(val - mx); __syncthreads();
  for (int o = 32; o > 0; o >>= 1) { if (j < o) red[j] += red[j + o]; __syncthreads(); }
  if (j == 0) llh[b] = score_sh - (mx + __logf(red[0]));
}

// ---------- final: out = sum(llh)/sum(ntok) ----------
__global__ void final_reduce_kernel(const float* __restrict__ llh,
                                    const float* __restrict__ ntok,
                                    float* __restrict__ out)
{
  __shared__ float s1[BB], s2[BB];
  const int t = threadIdx.x;
  s1[t] = llh[t]; s2[t] = ntok[t];
  __syncthreads();
  for (int o = 16; o > 0; o >>= 1) {
    if (t < o) { s1[t] += s1[t + o]; s2[t] += s2[t + o]; }
    __syncthreads();
  }
  if (t == 0) out[0] = s1[0] / s2[0];
}

extern "C" void kernel_launch(void* const* d_in, const int* in_sizes, int n_in,
                              void* d_out, int out_size, void* d_ws, size_t ws_size,
                              hipStream_t stream)
{
  (void)in_sizes; (void)n_in; (void)out_size; (void)ws_size;
  const float* x       = (const float*)d_in[0];
  const int*   mask    = (const int*)  d_in[1];
  const int*   tags    = (const int*)  d_in[2];
  const float* Wih_f   = (const float*)d_in[3];
  const float* Whh_f   = (const float*)d_in[4];
  const float* bih_f   = (const float*)d_in[5];
  const float* bhh_f   = (const float*)d_in[6];
  const float* Wih_b   = (const float*)d_in[7];
  const float* Whh_b   = (const float*)d_in[8];
  const float* bih_b   = (const float*)d_in[9];
  const float* bhh_b   = (const float*)d_in[10];
  const float* W1      = (const float*)d_in[11];
  const float* b1      = (const float*)d_in[12];
  const float* W2      = (const float*)d_in[13];
  const float* b2      = (const float*)d_in[14];
  const float* start_t = (const float*)d_in[15];
  const float* end_t   = (const float*)d_in[16];
  const float* trans   = (const float*)d_in[17];
  float* out = (float*)d_out;

  // ---- carve workspace ----
  char* p = (char*)d_ws;
  auto carve = [&](size_t bytes) -> void* {
    void* r = (void*)p; p += (bytes + 255) & ~(size_t)255; return r;
  };
  const int MROWS = BB * SS;                          // 16384
  bf16* xb    = (bf16*)carve((size_t)MROWS * DD * 2); // x in bf16
  bf16* wihfb = (bf16*)carve((size_t)NG * DD * 2);
  bf16* wihbb = (bf16*)carve((size_t)NG * DD * 2);
  bf16* whhfb = (bf16*)carve((size_t)NG * HH * 2);
  bf16* whhbb = (bf16*)carve((size_t)NG * HH * 2);
  bf16* w1b   = (bf16*)carve((size_t)HH * (2 * HH) * 2);
  bf16* w2b   = (bf16*)carve((size_t)TT * HH * 2);
  bf16* pre_f = (bf16*)carve((size_t)MROWS * NG * 2);
  bf16* pre_b = (bf16*)carve((size_t)MROWS * NG * 2);
  bf16* hcat  = (bf16*)carve((size_t)MROWS * (2 * HH) * 2);
  bf16* x1    = (bf16*)carve((size_t)MROWS * HH * 2);
  float* emis = (float*)carve((size_t)MROWS * TT * 4);
  float* llh  = (float*)carve((size_t)BB * 4);
  float* ntok = (float*)carve((size_t)BB * 4);

  auto cdiv = [](int a, int b) { return (a + b - 1) / b; };

  // ---- f32 -> bf16 conversions ----
  const int nX = MROWS * DD;
  f32_to_bf16_kernel<<<cdiv(nX, 256), 256, 0, stream>>>(x, xb, nX);
  f32_to_bf16_kernel<<<cdiv(NG * DD, 256), 256, 0, stream>>>(Wih_f, wihfb, NG * DD);
  f32_to_bf16_kernel<<<cdiv(NG * DD, 256), 256, 0, stream>>>(Wih_b, wihbb, NG * DD);
  f32_to_bf16_kernel<<<cdiv(NG * HH, 256), 256, 0, stream>>>(Whh_f, whhfb, NG * HH);
  f32_to_bf16_kernel<<<cdiv(NG * HH, 256), 256, 0, stream>>>(Whh_b, whhbb, NG * HH);
  f32_to_bf16_kernel<<<cdiv(HH * 2 * HH, 256), 256, 0, stream>>>(W1, w1b, HH * 2 * HH);
  f32_to_bf16_kernel<<<cdiv(TT * HH, 256), 256, 0, stream>>>(W2, w2b, TT * HH);

  // ---- pre-activations: pre = x @ Wih^T + (bih + bhh), both directions ----
  gemm_bf16_wmma<1, 0><<<dim3(NG / 16, MROWS / 64), 128, 0, stream>>>(
      xb, wihfb, bih_f, bhh_f, pre_f, MROWS, NG, DD);
  gemm_bf16_wmma<1, 0><<<dim3(NG / 16, MROWS / 64), 128, 0, stream>>>(
      xb, wihbb, bih_b, bhh_b, pre_b, MROWS, NG, DD);

  // ---- BiLSTM recurrence (both directions in one launch) ----
  lstm_recurrence<<<4, 1024, 0, stream>>>(pre_f, pre_b, whhfb, whhbb, hcat);

  // ---- x1 = hcat @ W1^T + b1 ----
  gemm_bf16_wmma<1, 0><<<dim3(HH / 16, MROWS / 64), 128, 0, stream>>>(
      hcat, w1b, b1, nullptr, x1, MROWS, HH, 2 * HH);

  // ---- emissions = gelu(x1 @ W2^T + b2), f32 output ----
  gemm_bf16_wmma<0, 1><<<dim3(TT / 16, MROWS / 64), 128, 0, stream>>>(
      x1, w2b, b2, nullptr, emis, MROWS, TT, HH);

  // ---- CRF per-batch llh + final reduce ----
  crf_llh_kernel<<<BB, TT, 0, stream>>>(emis, tags, mask, start_t, end_t, trans, llh, ntok);
  final_reduce_kernel<<<1, BB, 0, stream>>>(llh, ntok, out);
}
